// TransformerBlock_31688268710612
// MI455X (gfx1250) — compile-verified, matrix operand order checked by reading, not run on hardware
//
#include <hip/hip_runtime.h>

// ---------------- problem constants ----------------
#define B_    4096
#define S_    9
#define P_    4
#define D_    141
#define DFF_  1024
#define MROWS (B_ * P_)        // 16384 rows for all GEMMs
#define DPAD  160              // K-pad for D (mult of 32)
#define NBLK  192              // per-matrix N pad inside qkv stack (mult of 64)
#define NQKV  (3 * NBLK)       // 576
#define NG    (4 * DFF_)       // 4096 (I,F,O,Z stacked)

// GEMM tiling
#define TM    256              // block tile M
#define TN    64               // block tile N
#define KC    32               // K chunk per WMMA
#define PITCH 40               // LDS row pitch in bf16 elems (80B, breaks bank conflicts)

typedef __bf16 bf16t;
typedef bf16t v16bf __attribute__((ext_vector_type(16)));
typedef float v8f   __attribute__((ext_vector_type(8)));

// ---------------- helpers ----------------
static __device__ __forceinline__ unsigned short f2bf(float f) {
  unsigned u = __float_as_uint(f);
  unsigned r = 0x7FFFu + ((u >> 16) & 1u);   // round-to-nearest-even
  u += r;
  return (unsigned short)(u >> 16);
}

// low 32 bits of a generic pointer to __shared__ == LDS byte offset
static __device__ __forceinline__ unsigned lds_addr(const void* p) {
  return (unsigned)(uintptr_t)p;
}

// async DMA: 16 bytes global -> LDS, tracked by ASYNCcnt
static __device__ __forceinline__ void async_copy16(unsigned lds, const void* g) {
#if defined(__HIP_DEVICE_COMPILE__)
  asm volatile("global_load_async_to_lds_b128 %0, %1, off"
               :: "v"(lds), "v"((unsigned long long)(uintptr_t)g)
               : "memory");
#endif
}
static __device__ __forceinline__ void wait_async0() {
#if defined(__HIP_DEVICE_COMPILE__)
  asm volatile("s_wait_asynccnt 0x0" ::: "memory");
#endif
}

// Load one 16x32 bf16 fragment row for this lane:
//   VGPR0..3 : K = half*8 + 0..7 ; VGPR4..7 : K = 16 + half*8 + 0..7
static __device__ __forceinline__ v16bf load_frag(const unsigned short* rowp, int half) {
  union { uint4 u[2]; v16bf v; } U;
  U.u[0] = *reinterpret_cast<const uint4*>(rowp + half * 8);
  U.u[1] = *reinterpret_cast<const uint4*>(rowp + 16 + half * 8);
  return U.v;
}

static __device__ __forceinline__ v8f wmma_bf16(v16bf a, v16bf b, v8f c) {
#if defined(__HIP_DEVICE_COMPILE__)
  return __builtin_amdgcn_wmma_f32_16x16x32_bf16(false, a, false, b,
                                                 (short)0, c, false, false);
#else
  return c;
#endif
}

// ---------------- GEMM: C[M x N] (+)= A[M x K]bf16 @ Bt[N x K]bf16^T + bias ----------------
// grid = (M/256, N/64), block = 256 threads = 8 waves.
// Wave w computes rows [blkM + 32w, +32) x cols [blkN, blkN+64).
// A/B tiles are staged to LDS with async DMA, double buffered.
__global__ __launch_bounds__(256) void gemm_bf16_wmma(
    const unsigned short* __restrict__ A, int lda,
    const unsigned short* __restrict__ Bt, int ldb,
    float* __restrict__ C, int ldc,
    const float* __restrict__ bias, int K, int accumulate) {
  __shared__ unsigned short ldsA[2][TM * PITCH];   // 2 * 256*40*2B = 40 KB
  __shared__ unsigned short ldsB[2][TN * PITCH];   // 2 *  64*40*2B = 10 KB

  const int tid  = threadIdx.x;
  const int lane = tid & 31;
  const int wave = tid >> 5;
  const int half = lane >> 4;
  const int ln16 = lane & 15;
  const int blkM = blockIdx.x * TM;
  const int blkN = blockIdx.y * TN;

  // async-issue helper: fill buffer `buf` with K-chunk starting at k0
  auto issue = [&](int buf, int k0) {
    // A tile: 256 rows x 32 cols -> 1024 x 16B chunks, 4 per thread
#pragma unroll
    for (int i = 0; i < 4; ++i) {
      int chunk = tid + i * 256;
      int row = chunk >> 2, cc = chunk & 3;
      const unsigned short* g = A + (size_t)(blkM + row) * lda + k0 + cc * 8;
      async_copy16(lds_addr(&ldsA[buf][row * PITCH + cc * 8]), g);
    }
    // B tile: 64 rows x 32 cols -> 256 x 16B chunks, 1 per thread
    {
      int row = tid >> 2, cc = tid & 3;
      const unsigned short* g = Bt + (size_t)(blkN + row) * ldb + k0 + cc * 8;
      async_copy16(lds_addr(&ldsB[buf][row * PITCH + cc * 8]), g);
    }
  };

  v8f acc[2][4];
#pragma unroll
  for (int mf = 0; mf < 2; ++mf)
#pragma unroll
    for (int nf = 0; nf < 4; ++nf)
      acc[mf][nf] = (v8f){0.f,0.f,0.f,0.f,0.f,0.f,0.f,0.f};

  const int nk = K / KC;
  issue(0, 0);
  for (int kc = 0; kc < nk; ++kc) {
    const int buf = kc & 1;
    wait_async0();        // own async writes for current (and any prior) chunk done
    __syncthreads();      // everyone's writes for this chunk are visible
    if (kc + 1 < nk) issue(buf ^ 1, (kc + 1) * KC);   // overlap DMA with compute

    v16bf af[2];
#pragma unroll
    for (int mf = 0; mf < 2; ++mf)
      af[mf] = load_frag(&ldsA[buf][(wave * 32 + mf * 16 + ln16) * PITCH], half);
#pragma unroll
    for (int nf = 0; nf < 4; ++nf) {
      v16bf bfr = load_frag(&ldsB[buf][(nf * 16 + ln16) * PITCH], half);
#pragma unroll
      for (int mf = 0; mf < 2; ++mf)
        acc[mf][nf] = wmma_bf16(af[mf], bfr, acc[mf][nf]);
    }
  }

  // epilogue: C/D layout -> lane holds N = ln16, rows (half*8 + r)
#pragma unroll
  for (int nf = 0; nf < 4; ++nf) {
    const int n = blkN + nf * 16 + ln16;
    const float bv = bias ? bias[n] : 0.f;
#pragma unroll
    for (int mf = 0; mf < 2; ++mf) {
#pragma unroll
      for (int r = 0; r < 8; ++r) {
        const int m = blkM + wave * 32 + mf * 16 + half * 8 + r;
        float* cp = C + (size_t)m * ldc + n;
        const float v = acc[mf][nf][r] + bv;
        if (accumulate) *cp += v; else *cp = v;
      }
    }
  }
}

// ---------------- f32 -> bf16 weight conversion with column padding ----------------
__global__ void conv_w(const float* __restrict__ src, unsigned short* __restrict__ dst,
                       int rows, int cols, int ld, int rowOff) {
  int idx = blockIdx.x * blockDim.x + threadIdx.x;
  if (idx >= rows * ld) return;
  int r = idx / ld, c = idx - r * ld;
  dst[(size_t)(rowOff + r) * ld + c] = (c < cols) ? f2bf(src[(size_t)r * cols + c]) : 0;
}

// ---------------- per-step s_t -> bf16 padded [MROWS x DPAD] ----------------
__global__ void conv_s(const float* __restrict__ state, unsigned short* __restrict__ Sbf, int t) {
  int idx = blockIdx.x * blockDim.x + threadIdx.x;
  if (idx >= MROWS * DPAD) return;
  int row = idx / DPAD, d = idx - row * DPAD;
  int b = row >> 2, p = row & 3;
  float v = 0.f;
  if (d < D_) v = state[(((size_t)b * S_ + t) * P_ + p) * D_ + d];
  Sbf[idx] = f2bf(v);
}

// ---------------- attention + layernorm (one block per batch) ----------------
__global__ __launch_bounds__(64) void attn_ln(
    const float* __restrict__ sproj, const float* __restrict__ hproj,
    const float* __restrict__ state, const float* __restrict__ n1g,
    const float* __restrict__ n1b, float* __restrict__ Aout,
    unsigned short* __restrict__ Z2bf, int t) {
  const int b = blockIdx.x, tid = threadIdx.x;
  __shared__ float q[P_][D_], k[P_][D_], v[P_][D_], z2[P_][D_];
  __shared__ float A[P_][P_], mu[P_], rs[P_];

  for (int idx = tid; idx < P_ * D_; idx += 64) {
    int p = idx / D_, d = idx - p * D_;
    size_t row = (size_t)(b * P_ + p) * NQKV;
    q[p][d] = sproj[row + d]           * hproj[row + d];
    k[p][d] = sproj[row + NBLK + d]    * hproj[row + NBLK + d];
    v[p][d] = sproj[row + 2*NBLK + d]  * hproj[row + 2*NBLK + d];
  }
  __syncthreads();

  if (tid < 16) {
    int p = tid >> 2, qq = tid & 3;
    float s = 0.f;
    for (int d = 0; d < D_; ++d) s += q[p][d] * k[qq][d];
    A[p][qq] = s * rsqrtf((float)D_);
  }
  __syncthreads();
  if (tid < P_) {
    int p = tid;
    float mx = fmaxf(fmaxf(A[p][0], A[p][1]), fmaxf(A[p][2], A[p][3]));
    float e0 = __expf(A[p][0]-mx), e1 = __expf(A[p][1]-mx),
          e2 = __expf(A[p][2]-mx), e3 = __expf(A[p][3]-mx);
    float inv = 1.f / (e0 + e1 + e2 + e3);
    A[p][0] = e0*inv; A[p][1] = e1*inv; A[p][2] = e2*inv; A[p][3] = e3*inv;
  }
  __syncthreads();

  for (int idx = tid; idx < P_ * D_; idx += 64) {
    int p = idx / D_, d = idx - p * D_;
    float a = A[p][0]*v[0][d] + A[p][1]*v[1][d] + A[p][2]*v[2][d] + A[p][3]*v[3][d];
    float si = state[(((size_t)b * S_ + t) * P_ + p) * D_ + d];
    z2[p][d] = si + a;
  }
  __syncthreads();
  if (tid < P_) {
    int p = tid;
    float m = 0.f;
    for (int d = 0; d < D_; ++d) m += z2[p][d];
    m /= (float)D_;
    float var = 0.f;
    for (int d = 0; d < D_; ++d) { float x = z2[p][d] - m; var += x * x; }
    var /= (float)D_;
    mu[p] = m; rs[p] = rsqrtf(var + 1e-5f);
  }
  __syncthreads();

  for (int idx = tid; idx < P_ * DPAD; idx += 64) {
    int p = idx / DPAD, d = idx - p * DPAD;
    float val = 0.f;
    if (d < D_) val = (z2[p][d] - mu[p]) * rs[p] * n1g[d] + n1b[d];
    Z2bf[(size_t)(b * P_ + p) * DPAD + d] = f2bf(val);
  }
  if (tid < 16)  // A_tensor[b, t, 0, p, q]
    Aout[((size_t)b * S_ + t) * 16 + tid] = A[tid >> 2][tid & 3];
}

// ---------------- mLSTM state update ----------------
__global__ void state_update(const float* __restrict__ G, const float* __restrict__ mvec,
                             float* __restrict__ Cst, float* __restrict__ Mst,
                             float* __restrict__ Nst, float* __restrict__ Hst,
                             unsigned short* __restrict__ Hbf, float* __restrict__ Hout, int t) {
  int idx = blockIdx.x * blockDim.x + threadIdx.x;
  if (idx >= MROWS * DFF_) return;
  int row = idx >> 10, col = idx & 1023;
  int b = row >> 2, p = row & 3;
  size_t g = (size_t)row * NG + col;
  float I = G[g], F = G[g + DFF_], O = G[g + 2*DFF_], Z = G[g + 3*DFF_];
  float Mo = Mst[idx], No = Nst[idx], Co = Cst[idx], Ho = Hst[idx];
  float mi = mvec[((size_t)b * S_ + t) * P_ + p];

  float Mn = fmaxf(F + Mo, I);
  float Ig = __expf(I - Mn);
  float Fg = __expf(F + Mo - Mn);
  float Og = 1.f / (1.f + __expf(-O));
  float Nn = Fg * No + Ig;
  float Zg = tanhf(Z);
  float Cn = (Co * Fg + Zg * Ig) * mi + (1.f - mi) * Co;
  float Hn = Og * (Cn / Nn) * mi + (1.f - mi) * Ho;

  Cst[idx] = Cn; Mst[idx] = Mn; Nst[idx] = Nn; Hst[idx] = Hn;
  Hbf[idx] = f2bf(Hn);
  if (Hout) Hout[idx] = Hn;   // [B,P,DFF] == row*DFF + col
}

// ---------------- host orchestration ----------------
extern "C" void kernel_launch(void* const* d_in, const int* in_sizes, int n_in,
                              void* d_out, int out_size, void* d_ws, size_t ws_size,
                              hipStream_t stream) {
  (void)in_sizes; (void)n_in; (void)out_size; (void)ws_size;
  const float* state = (const float*)d_in[0];
  const float* mvec  = (const float*)d_in[1];
  const float* qw = (const float*)d_in[2],  * qb = (const float*)d_in[3];
  const float* kw = (const float*)d_in[4],  * kb = (const float*)d_in[5];
  const float* vw = (const float*)d_in[6],  * vb = (const float*)d_in[7];
  const float* cqw = (const float*)d_in[8],  * cqb = (const float*)d_in[9];
  const float* ckw = (const float*)d_in[10], * ckb = (const float*)d_in[11];
  const float* cvw = (const float*)d_in[12], * cvb = (const float*)d_in[13];
  const float* n1g = (const float*)d_in[14], * n1b = (const float*)d_in[15];
  const float* wi_w = (const float*)d_in[16], * wi_b = (const float*)d_in[17];
  const float* wf_w = (const float*)d_in[18], * wf_b = (const float*)d_in[19];
  const float* wo_w = (const float*)d_in[20], * wo_b = (const float*)d_in[21];
  const float* wz_w = (const float*)d_in[22], * wz_b = (const float*)d_in[23];
  const float* ri_w = (const float*)d_in[24], * ri_b = (const float*)d_in[25];
  const float* rf_w = (const float*)d_in[26], * rf_b = (const float*)d_in[27];
  const float* ro_w = (const float*)d_in[28], * ro_b = (const float*)d_in[29];
  const float* rz_w = (const float*)d_in[30], * rz_b = (const float*)d_in[31];

  float* outH = (float*)d_out;                       // [B,P,DFF]
  float* outA = outH + (size_t)MROWS * DFF_;         // [B,S,1,P,P]

  // --- workspace bump allocator (256B aligned) ---
  char* ws = (char*)d_ws; size_t off = 0;
  auto alloc = [&](size_t bytes) -> void* {
    void* p = ws + off; off = (off + bytes + 255) & ~(size_t)255; return p;
  };
  unsigned short* Wsqkv = (unsigned short*)alloc((size_t)NQKV * DPAD * 2);
  unsigned short* Wc    = (unsigned short*)alloc((size_t)NQKV * DFF_ * 2);
  unsigned short* Wg    = (unsigned short*)alloc((size_t)NG * DFF_ * 2);
  unsigned short* Rg    = (unsigned short*)alloc((size_t)NG * DPAD * 2);
  float* bsqkv = (float*)alloc(NQKV * 4);
  float* bc    = (float*)alloc(NQKV * 4);
  float* bg    = (float*)alloc(NG * 4);
  float* br    = (float*)alloc(NG * 4);
  unsigned short* Sbf  = (unsigned short*)alloc((size_t)MROWS * DPAD * 2);
  unsigned short* Hbf  = (unsigned short*)alloc((size_t)MROWS * DFF_ * 2);
  unsigned short* Z2bf = (unsigned short*)alloc((size_t)MROWS * DPAD * 2);
  float* sproj = (float*)alloc((size_t)MROWS * NQKV * 4);
  float* hproj = (float*)alloc((size_t)MROWS * NQKV * 4);
  float* G     = (float*)alloc((size_t)MROWS * NG * 4);
  float* Cst   = (float*)alloc((size_t)MROWS * DFF_ * 4);
  float* Mst   = (float*)alloc((size_t)MROWS * DFF_ * 4);
  float* Nst   = (float*)alloc((size_t)MROWS * DFF_ * 4);
  float* Hst   = (float*)alloc((size_t)MROWS * DFF_ * 4);

  // --- zero init (states + padded weight stacks + Hbf) ---
  hipMemsetAsync(Wsqkv, 0, (size_t)NQKV * DPAD * 2, stream);
  hipMemsetAsync(bsqkv, 0, NQKV * 4, stream);
  hipMemsetAsync(bc,    0, NQKV * 4, stream);
  hipMemsetAsync(Hbf,   0, (size_t)MROWS * DFF_ * 2, stream);
  hipMemsetAsync(Cst,   0, (size_t)MROWS * DFF_ * 4, stream);
  hipMemsetAsync(Mst,   0, (size_t)MROWS * DFF_ * 4, stream);
  hipMemsetAsync(Nst,   0, (size_t)MROWS * DFF_ * 4, stream);
  hipMemsetAsync(Hst,   0, (size_t)MROWS * DFF_ * 4, stream);

  // --- weight conversion to bf16 (padded / stacked) ---
  auto cw = [&](const float* src, unsigned short* dst, int rows, int cols, int ld, int rowOff) {
    int n = rows * ld;
    conv_w<<<(n + 255) / 256, 256, 0, stream>>>(src, dst, rows, cols, ld, rowOff);
  };
  cw(qw,  Wsqkv, D_, D_, DPAD, 0);        // q rows [0,141)
  cw(kw,  Wsqkv, D_, D_, DPAD, NBLK);     // k rows [192,333)
  cw(vw,  Wsqkv, D_, D_, DPAD, 2*NBLK);   // v rows [384,525)
  cw(cqw, Wc, D_, DFF_, DFF_, 0);
  cw(ckw, Wc, D_, DFF_, DFF_, NBLK);
  cw(cvw, Wc, D_, DFF_, DFF_, 2*NBLK);
  cw(wi_w, Wg, DFF_, DFF_, DFF_, 0);
  cw(wf_w, Wg, DFF_, DFF_, DFF_, DFF_);
  cw(wo_w, Wg, DFF_, DFF_, DFF_, 2*DFF_);
  cw(wz_w, Wg, DFF_, DFF_, DFF_, 3*DFF_);
  cw(ri_w, Rg, DFF_, D_, DPAD, 0);
  cw(rf_w, Rg, DFF_, D_, DPAD, DFF_);
  cw(ro_w, Rg, DFF_, D_, DPAD, 2*DFF_);
  cw(rz_w, Rg, DFF_, D_, DPAD, 3*DFF_);
  // zero filler rows of Wc (rows [141,192)+k*192 stay zero so padded hproj cols are 0)
  hipMemsetAsync(Wc + (size_t)D_ * DFF_, 0, (size_t)(NBLK - D_) * DFF_ * 2, stream);
  hipMemsetAsync(Wc + (size_t)(NBLK + D_) * DFF_, 0, (size_t)(NBLK - D_) * DFF_ * 2, stream);
  hipMemsetAsync(Wc + (size_t)(2*NBLK + D_) * DFF_, 0, (size_t)(NBLK - D_) * DFF_ * 2, stream);

  // --- bias stacks ---
  hipMemcpyAsync(bsqkv,            qb, D_ * 4, hipMemcpyDeviceToDevice, stream);
  hipMemcpyAsync(bsqkv + NBLK,     kb, D_ * 4, hipMemcpyDeviceToDevice, stream);
  hipMemcpyAsync(bsqkv + 2*NBLK,   vb, D_ * 4, hipMemcpyDeviceToDevice, stream);
  hipMemcpyAsync(bc,               cqb, D_ * 4, hipMemcpyDeviceToDevice, stream);
  hipMemcpyAsync(bc + NBLK,        ckb, D_ * 4, hipMemcpyDeviceToDevice, stream);
  hipMemcpyAsync(bc + 2*NBLK,      cvb, D_ * 4, hipMemcpyDeviceToDevice, stream);
  hipMemcpyAsync(bg,               wi_b, DFF_ * 4, hipMemcpyDeviceToDevice, stream);
  hipMemcpyAsync(bg + DFF_,        wf_b, DFF_ * 4, hipMemcpyDeviceToDevice, stream);
  hipMemcpyAsync(bg + 2*DFF_,      wo_b, DFF_ * 4, hipMemcpyDeviceToDevice, stream);
  hipMemcpyAsync(bg + 3*DFF_,      wz_b, DFF_ * 4, hipMemcpyDeviceToDevice, stream);
  hipMemcpyAsync(br,               ri_b, DFF_ * 4, hipMemcpyDeviceToDevice, stream);
  hipMemcpyAsync(br + DFF_,        rf_b, DFF_ * 4, hipMemcpyDeviceToDevice, stream);
  hipMemcpyAsync(br + 2*DFF_,      ro_b, DFF_ * 4, hipMemcpyDeviceToDevice, stream);
  hipMemcpyAsync(br + 3*DFF_,      rz_b, DFF_ * 4, hipMemcpyDeviceToDevice, stream);

  // --- sequential scan over S steps ---
  const dim3 blk(256);
  for (int t = 0; t < S_; ++t) {
    { int n = MROWS * DPAD;
      conv_s<<<(n + 255) / 256, 256, 0, stream>>>(state, Sbf, t); }

    // sproj = Sbf @ Wsqkv^T + bsqkv        (M=16384, N=576, K=160)
    gemm_bf16_wmma<<<dim3(MROWS / TM, NQKV / TN), blk, 0, stream>>>(
        Sbf, DPAD, Wsqkv, DPAD, sproj, NQKV, bsqkv, DPAD, 0);
    // hproj = Hbf @ Wc^T + bc              (M=16384, N=576, K=1024)
    gemm_bf16_wmma<<<dim3(MROWS / TM, NQKV / TN), blk, 0, stream>>>(
        Hbf, DFF_, Wc, DFF_, hproj, NQKV, bc, DFF_, 0);

    attn_ln<<<B_, 64, 0, stream>>>(sproj, hproj, state, n1g, n1b, outA, Z2bf, t);

    // G  = Hbf  @ Wg^T + bg                (M=16384, N=4096, K=1024)
    gemm_bf16_wmma<<<dim3(MROWS / TM, NG / TN), blk, 0, stream>>>(
        Hbf, DFF_, Wg, DFF_, G, NG, bg, DFF_, 0);
    // G += Z2bf @ Rg^T + br                (M=16384, N=4096, K=160)
    gemm_bf16_wmma<<<dim3(MROWS / TM, NG / TN), blk, 0, stream>>>(
        Z2bf, DPAD, Rg, DPAD, G, NG, br, DPAD, 1);

    { int n = MROWS * DFF_;
      state_update<<<(n + 255) / 256, 256, 0, stream>>>(
          G, mvec, Cst, Mst, Nst, Hst, Hbf,
          (t == S_ - 1) ? outH : (float*)nullptr, t); }
  }
}
